// SparseMixtureOfExperts_60868276519715
// MI455X (gfx1250) — matrix operand order repaired and checked by reading, not compiled, not run on hardware
//
#include <hip/hip_runtime.h>
#include <hip/hip_bf16.h>
#include <math.h>

// ---------------------------------------------------------------------------
// Sparse MoE (top-2 of 16 experts), SwiGLU expert FFN + LayerNorm, f16 WMMA.
// B=4, N=2048 -> T=8192 tokens, D=1024, F=4096, E=16, K=2.
// Router gathers tokens per expert; fused expert kernel runs f16 WMMA with
// f32 accumulate. Pre-pass converts weights to f16 AND transposes them
// K-major so B fragments are 32B contiguous loads (2x global_load_b128).
// X is pre-converted to f16 and staged into LDS with CDNA5 async copies.
// ---------------------------------------------------------------------------

constexpr int TT = 8192;   // tokens
constexpr int DD = 1024;   // model dim
constexpr int FF = 4096;   // ffn dim
constexpr int EE = 16;     // experts
constexpr float EPS = 1e-5f;

typedef _Float16 half16 __attribute__((ext_vector_type(16)));
typedef _Float16 half8  __attribute__((ext_vector_type(8)));
typedef float    float8 __attribute__((ext_vector_type(8)));

// LDS strides padded so 16 consecutive rows hit distinct banks, 16B aligned.
constexpr int XSTR = DD + 8;   // halfs; 2064B/row -> 516 dwords, 516%64=4
constexpr int HSTR = 128 + 8;  // halfs; 272B/row  -> 68 dwords,  68%64=4

// ---------------------------------------------------------------------------
// WMMA fragment helpers (wave32; this code's contiguous k order, not the ISA table order)
// A 16x32 f16: lane holds row M=lane&15; kb=16*(lane>>4);
//   element j -> K=kb+j (16 contiguous halves)
// B 32x16 f16: lane holds col N=lane&15; element j -> K=16*(lane>>4)+j
//   (with K-major transposed weights this is 16 contiguous halves)
// C/D 16x16 f32: VGPR r, lane l -> M = r + 8*(l>>4), N = l&15
// ---------------------------------------------------------------------------
__device__ inline half16 load_a_lds(const _Float16* __restrict__ row, int k0, int kb) {
  const half8 lo = *(const half8*)(row + k0 + kb);        // ds_load_b128
  const half8 hi = *(const half8*)(row + k0 + kb + 8);
  return __builtin_shufflevector(lo, hi, 0, 1, 2, 3, 4, 5, 6, 7,
                                 8, 9, 10, 11, 12, 13, 14, 15);
}

// fallback path: column gather from row-major f32 weights
__device__ inline half16 load_b_col(const float* __restrict__ W, long rowBase,
                                    int ld, int n, int kb) {
  half16 b;
#pragma unroll
  for (int j = 0; j < 16; ++j)
    b[j] = (_Float16)W[(rowBase + kb + j) * (long)ld + n];
  return b;
}

__device__ inline float8 wmma_f16(half16 a, half16 b, float8 c) {
  return __builtin_amdgcn_wmma_f32_16x16x32_f16(
      false, a, false, b, (short)0, c, false, false);
}

// CDNA5 async global->LDS copy (tracked with ASYNCcnt)
__device__ inline void async_copy_b128(unsigned lds_off, unsigned long long gaddr) {
  asm volatile("global_load_async_to_lds_b128 %0, %1, off"
               :: "v"(lds_off), "v"(gaddr) : "memory");
}
__device__ inline void wait_asynccnt0() {
  asm volatile("s_wait_asynccnt 0x0" ::: "memory");
}

// ---------------------------------------------------------------------------
// f32 -> f16 flat conversion (n multiple of 8)
// ---------------------------------------------------------------------------
__global__ __launch_bounds__(256) void cvt_f32_f16(const float* __restrict__ src,
                                                   _Float16* __restrict__ dst,
                                                   long n) {
  const long i = ((long)blockIdx.x * 256 + threadIdx.x) * 8;
  if (i + 8 > n) return;
  const float4 a = ((const float4*)(src + i))[0];
  const float4 b = ((const float4*)(src + i))[1];
  half8 o;
  o[0] = (_Float16)a.x; o[1] = (_Float16)a.y;
  o[2] = (_Float16)a.z; o[3] = (_Float16)a.w;
  o[4] = (_Float16)b.x; o[5] = (_Float16)b.y;
  o[6] = (_Float16)b.z; o[7] = (_Float16)b.w;
  *(half8*)(dst + i) = o;
}

// ---------------------------------------------------------------------------
// f32 -> f16 transpose: src [E][R][C] -> dst [E][C][R], 32x32 LDS tiles
// ---------------------------------------------------------------------------
__global__ __launch_bounds__(256) void cvt_tr_f16(const float* __restrict__ src,
                                                  _Float16* __restrict__ dst,
                                                  int R, int C) {
  __shared__ _Float16 tile[32][33];
  const size_t mat = (size_t)R * C;
  const float* S = src + (size_t)blockIdx.z * mat;
  _Float16* Dst = dst + (size_t)blockIdx.z * mat;
  const int c0 = blockIdx.x * 32, r0 = blockIdx.y * 32;
  const int tx = threadIdx.x & 31;
  const int ty = threadIdx.x >> 5;   // 0..7
#pragma unroll
  for (int i = 0; i < 4; ++i) {
    const int r = ty + i * 8;
    tile[r][tx] = (_Float16)S[(size_t)(r0 + r) * C + (c0 + tx)];
  }
  __syncthreads();
#pragma unroll
  for (int i = 0; i < 4; ++i) {
    const int c = ty + i * 8;
    Dst[(size_t)(c0 + c) * R + (r0 + tx)] = tile[tx][c];
  }
}

// ---------------------------------------------------------------------------
// Kernel 1: router. One wave per token.
// ---------------------------------------------------------------------------
__global__ __launch_bounds__(256) void moe_router(
    const float* __restrict__ x, const float* __restrict__ Wr,
    const float* __restrict__ br, int* __restrict__ cnt,
    int* __restrict__ toks, float* __restrict__ twgt) {
  const int wave = threadIdx.x >> 5;
  const int lane = threadIdx.x & 31;
  const int t = blockIdx.x * 8 + wave;
  if (t >= TT) return;

  float acc[EE];
#pragma unroll
  for (int e = 0; e < EE; ++e) acc[e] = 0.f;

  const float* xp = x + (long)t * DD;
  for (int d = lane; d < DD; d += 32) {
    const float xv = xp[d];
    const float* wr = Wr + d * EE;
#pragma unroll
    for (int e = 0; e < EE; ++e) acc[e] += xv * wr[e];
  }
#pragma unroll
  for (int e = 0; e < EE; ++e) {
    float v = acc[e];
#pragma unroll
    for (int m = 16; m >= 1; m >>= 1) v += __shfl_xor(v, m, 32);
    acc[e] = v + br[e];
  }

  if (lane == 0) {
    float mx = acc[0];
#pragma unroll
    for (int e = 1; e < EE; ++e) mx = fmaxf(mx, acc[e]);
    float p[EE];
#pragma unroll
    for (int e = 0; e < EE; ++e) p[e] = __expf(acc[e] - mx);
    int i1 = 0; float v1 = p[0];
#pragma unroll
    for (int e = 1; e < EE; ++e) if (p[e] > v1) { v1 = p[e]; i1 = e; }
    int i2 = (i1 == 0) ? 1 : 0; float v2 = p[i2];
#pragma unroll
    for (int e = 0; e < EE; ++e)
      if (e != i1 && p[e] > v2) { v2 = p[e]; i2 = e; }
    const float inv = 1.f / (v1 + v2);   // softmax denom cancels in renorm
    int p1 = atomicAdd(&cnt[i1], 1);
    toks[i1 * TT + p1] = t; twgt[i1 * TT + p1] = v1 * inv;
    int p2 = atomicAdd(&cnt[i2], 1);
    toks[i2 * TT + p2] = t; twgt[i2 * TT + p2] = v2 * inv;
  }
}

// ---------------------------------------------------------------------------
// Kernel 2: fused expert FFN (templated on weight precision/layout).
// f16 path:  W1/W3 transposed [E][F][D], W2 transposed [E][D][F].
// f32 path:  original layouts, column-gather fragments.
// Block = 16 routed tokens of one expert, 256 threads = 8 waves.
// ---------------------------------------------------------------------------
template <typename WT>
__global__ __launch_bounds__(256) void moe_ffn_t(
    const float* __restrict__ x32, const _Float16* __restrict__ x16,
    const WT* __restrict__ W1, const WT* __restrict__ W3,
    const WT* __restrict__ W2,
    const float* __restrict__ gamma, const float* __restrict__ beta,
    const int* __restrict__ cnt, const int* __restrict__ toks,
    const float* __restrict__ twgt, float* __restrict__ out) {
  constexpr bool kF16 = __is_same(WT, _Float16);
  const int e = blockIdx.y;
  const int tile = blockIdx.x;
  const int nTok = cnt[e];
  if (tile * 16 >= nTok) return;

  __shared__ __align__(16) _Float16 Xs[16 * XSTR];
  __shared__ __align__(16) _Float16 Hs[16 * HSTR];
  __shared__ float redS[8][16];
  __shared__ float redS2[8][16];
  __shared__ int   tIdxS[16];
  __shared__ float twS[16];

  const int tid = threadIdx.x;
  if (tid < 16) {
    const int g = tile * 16 + tid;
    if (g < nTok) { tIdxS[tid] = toks[e * TT + g]; twS[tid] = twgt[e * TT + g]; }
    else          { tIdxS[tid] = 0;                twS[tid] = 0.f; }
  }
  __syncthreads();

  // stage X tile [16 x 1024] as f16 into padded LDS layout
  if constexpr (kF16) {
#pragma unroll
    for (int it = 0; it < 8; ++it) {
      const int c = tid + it * 256;          // 2048 chunks of 8 halfs
      const int m = c >> 7;                  // 128 chunks per row
      const int dd = (c & 127) * 8;
      const unsigned lds_off = (unsigned)(uintptr_t)&Xs[m * XSTR + dd];
      const unsigned long long ga =
          (unsigned long long)(uintptr_t)(x16 + (long)tIdxS[m] * DD + dd);
      async_copy_b128(lds_off, ga);
    }
    wait_asynccnt0();
  } else {
    for (int i = tid; i < 16 * DD; i += 256) {
      const int m = i >> 10, d = i & (DD - 1);
      Xs[m * XSTR + d] = (_Float16)x32[(long)tIdxS[m] * DD + d];
    }
  }
  __syncthreads();

  const int wave = tid >> 5;
  const int lane = tid & 31;
  const int hs = lane >> 4;      // half-wave select
  const int nn = lane & 15;
  const int kbA = hs * 16;
  const int kbB = hs * 16;
  const _Float16* xrow = &Xs[(lane & 15) * XSTR];
  const _Float16* hrow = &Hs[(lane & 15) * HSTR];

  float8 acc[8];
#pragma unroll
  for (int t = 0; t < 8; ++t) acc[t] = (float8){};

  for (int fc = 0; fc < FF; fc += 128) {
    // --- GEMM1: this wave's 16-wide h column tile, K = D = 1024 ---
    float8 a1 = (float8){}, a3 = (float8){};
    const int fcol = fc + wave * 16 + nn;
    // transposed rows (f16 path): W1t/W3t row 'fcol' holds D contiguous K
    const WT* b1row = W1 + ((long)e * FF + fcol) * DD;
    const WT* b3row = W3 + ((long)e * FF + fcol) * DD;
    for (int k0 = 0; k0 < DD; k0 += 32) {
      const half16 a = load_a_lds(xrow, k0, kbA);
      half16 b1, b3;
      if constexpr (kF16) {
        b1 = *(const half16*)(b1row + k0 + kbB);   // 32B contiguous
        b3 = *(const half16*)(b3row + k0 + kbB);
      } else {
        b1 = load_b_col(W1, (long)e * DD + k0, FF, fcol, kbB);
        b3 = load_b_col(W3, (long)e * DD + k0, FF, fcol, kbB);
      }
      a1 = wmma_f16(a, b1, a1);
      a3 = wmma_f16(a, b3, a3);
    }
    // silu(a1) * a3 -> Hs (f16); fast sigmoid via v_rcp_f32
#pragma unroll
    for (int r = 0; r < 8; ++r) {
      const float v1 = a1[r], v3 = a3[r];
      const float sig = __builtin_amdgcn_rcpf(1.f + __expf(-v1));
      Hs[(r + hs * 8) * HSTR + wave * 16 + nn] = (_Float16)(v1 * sig * v3);
    }
    __syncthreads();
    // --- GEMM2: y[:, wave*128 .. +128) += h_chunk @ W2_chunk ---
#pragma unroll
    for (int kk = 0; kk < 128; kk += 32) {
      const half16 a = load_a_lds(hrow, kk, kbA);
#pragma unroll
      for (int t = 0; t < 8; ++t) {
        const int n = wave * 128 + t * 16 + nn;
        half16 b;
        if constexpr (kF16) {
          b = *(const half16*)(W2 + ((long)e * DD + n) * FF + fc + kk + kbB);
        } else {
          b = load_b_col(W2, (long)e * FF + fc + kk, DD, n, kbB);
        }
        acc[t] = wmma_f16(a, b, acc[t]);
      }
    }
    __syncthreads();
  }

  // --- LayerNorm over D=1024 per row ---
  float s[8], s2[8];
#pragma unroll
  for (int r = 0; r < 8; ++r) {
    s[r] = 0.f; s2[r] = 0.f;
#pragma unroll
    for (int t = 0; t < 8; ++t) { const float v = acc[t][r]; s[r] += v; s2[r] += v * v; }
  }
#pragma unroll
  for (int m = 8; m >= 1; m >>= 1) {
#pragma unroll
    for (int r = 0; r < 8; ++r) {
      s[r]  += __shfl_xor(s[r],  m, 32);
      s2[r] += __shfl_xor(s2[r], m, 32);
    }
  }
  if (nn == 0) {
#pragma unroll
    for (int r = 0; r < 8; ++r) {
      redS[wave][hs * 8 + r]  = s[r];
      redS2[wave][hs * 8 + r] = s2[r];
    }
  }
  __syncthreads();

  float mu[8], rinv[8];
#pragma unroll
  for (int r = 0; r < 8; ++r) {
    const int M = hs * 8 + r;
    float S = 0.f, S2 = 0.f;
#pragma unroll
    for (int w = 0; w < 8; ++w) { S += redS[w][M]; S2 += redS2[w][M]; }
    const float m_ = S * (1.f / (float)DD);
    const float var = S2 * (1.f / (float)DD) - m_ * m_;
    mu[r] = m_;
    rinv[r] = rsqrtf(var + EPS);
  }

  // gamma/beta, weighted scatter-add (K=2 accumulation across experts)
#pragma unroll
  for (int t = 0; t < 8; ++t) {
    const int col = wave * 128 + t * 16 + nn;
    const float g = gamma[e * DD + col];
    const float bt = beta[e * DD + col];
#pragma unroll
    for (int r = 0; r < 8; ++r) {
      const int M = hs * 8 + r;
      const float yn = (acc[t][r] - mu[r]) * rinv[r] * g + bt;
      atomicAdd(&out[(long)tIdxS[M] * DD + col], twS[M] * yn);
    }
  }
}

// ---------------------------------------------------------------------------
extern "C" void kernel_launch(void* const* d_in, const int* in_sizes, int n_in,
                              void* d_out, int out_size, void* d_ws, size_t ws_size,
                              hipStream_t stream) {
  const float* x     = (const float*)d_in[0];
  const float* Wr    = (const float*)d_in[1];
  const float* br    = (const float*)d_in[2];
  const float* W1    = (const float*)d_in[3];
  const float* W3    = (const float*)d_in[4];
  const float* W2    = (const float*)d_in[5];
  const float* gamma = (const float*)d_in[6];
  const float* beta  = (const float*)d_in[7];
  float* out = (float*)d_out;

  // workspace layout
  const size_t nW = (size_t)EE * DD * FF;           // 67,108,864 elems per W
  const size_t off_tok = 256;
  const size_t off_tw  = off_tok + (size_t)EE * TT * 4;
  const size_t off_xh  = off_tw  + (size_t)EE * TT * 4;
  const size_t off_w1  = off_xh  + (size_t)TT * DD * 2;
  const size_t off_w3  = off_w1  + nW * 2;
  const size_t off_w2  = off_w3  + nW * 2;
  const size_t total   = off_w2  + nW * 2;

  int*   cntp = (int*)d_ws;
  int*   toks = (int*)((char*)d_ws + off_tok);
  float* twp  = (float*)((char*)d_ws + off_tw);

  hipMemsetAsync(d_ws, 0, 256, stream);
  hipMemsetAsync(d_out, 0, (size_t)out_size * sizeof(float), stream);

  moe_router<<<TT / 8, 256, 0, stream>>>(x, Wr, br, cntp, toks, twp);

  dim3 grid(TT / 16, EE);
  if (ws_size >= total) {
    _Float16* xh  = (_Float16*)((char*)d_ws + off_xh);
    _Float16* w1t = (_Float16*)((char*)d_ws + off_w1);
    _Float16* w3t = (_Float16*)((char*)d_ws + off_w3);
    _Float16* w2t = (_Float16*)((char*)d_ws + off_w2);
    const long nx = (long)TT * DD;
    cvt_f32_f16<<<(unsigned)(nx / 8 / 256), 256, 0, stream>>>(x, xh, nx);
    // W1/W3: [E][D][F] -> [E][F][D] ; W2: [E][F][D_model] -> [E][D_model][F]
    dim3 g13(FF / 32, DD / 32, EE);
    cvt_tr_f16<<<g13, 256, 0, stream>>>(W1, w1t, DD, FF);
    cvt_tr_f16<<<g13, 256, 0, stream>>>(W3, w3t, DD, FF);
    dim3 g2(DD / 32, FF / 32, EE);
    cvt_tr_f16<<<g2, 256, 0, stream>>>(W2, w2t, FF, DD);
    moe_ffn_t<_Float16><<<grid, 256, 0, stream>>>(
        x, xh, w1t, w3t, w2t, gamma, beta, cntp, toks, twp, out);
  } else {
    moe_ffn_t<float><<<grid, 256, 0, stream>>>(
        x, nullptr, W1, W3, W2, gamma, beta, cntp, toks, twp, out);
  }
}
